// SCOT_16673063043322
// MI455X (gfx1250) — matrix operand order repaired and checked by reading, not compiled, not run on hardware
//
#include <hip/hip_runtime.h>

typedef _Float16 v16h __attribute__((ext_vector_type(16)));
typedef _Float16 v8h  __attribute__((ext_vector_type(8)));
typedef float    v8f  __attribute__((ext_vector_type(8)));
typedef int      v4i  __attribute__((ext_vector_type(4)));

#define WMMA_F16(a, b, c) \
  __builtin_amdgcn_wmma_f32_16x16x32_f16(false, (a), false, (b), (short)0, (c), false, false)

static __device__ __forceinline__ v16h cat8(v8h lo, v8h hi) {
  return __builtin_shufflevector(lo, hi, 0,1,2,3,4,5,6,7,8,9,10,11,12,13,14,15);
}

// ---- optional gfx1250 async global->LDS staging --------------------------
#if defined(__has_builtin)
#if __has_builtin(__builtin_amdgcn_global_load_async_to_lds_b128)
#define HAVE_ASYNC_LDS 1
#endif
#endif
#ifndef HAVE_ASYNC_LDS
#define HAVE_ASYNC_LDS 0
#endif

#if HAVE_ASYNC_LDS
typedef __attribute__((address_space(1))) v4i* gas4_t;  // global int4*
typedef __attribute__((address_space(3))) v4i* las4_t;  // LDS int4*
static __device__ __forceinline__ void wait_async0() {
#if __has_builtin(__builtin_amdgcn_s_wait_asynccnt)
  __builtin_amdgcn_s_wait_asynccnt(0);
#else
  asm volatile("s_wait_asynccnt 0x0" ::: "memory");
#endif
}
#endif

// Problem constants
#define BB   256
#define NS   30
#define SS   10
#define SQ   300      // S = NS*SS queries per batch
#define DD   256
#define HH   4
#define DH   64
#define LP   320      // padded key count (L=301 -> 320, divisible by 32)

// ---------------------------------------------------------------------------
// f32 -> f16 convert (grid-stride)
// ---------------------------------------------------------------------------
__global__ void cvt_f32_f16(const float* __restrict__ x, _Float16* __restrict__ y, int n) {
  for (int i = blockIdx.x * blockDim.x + threadIdx.x; i < n; i += gridDim.x * blockDim.x)
    y[i] = (_Float16)x[i];
}

// ---------------------------------------------------------------------------
// Prep: per batch, compact clicked keys to front (stable), gather user token,
// emit f16 key matrix [B, LP, D] and orig_pos [B, LP] (INT_MAX = invalid).
// ---------------------------------------------------------------------------
__global__ __launch_bounds__(320) void prep_kernel(
    const float* __restrict__ item, const float* __restrict__ user,
    const int* __restrict__ resp, const int* __restrict__ len,
    _Float16* __restrict__ keys16, int* __restrict__ orig) {
  const int b = blockIdx.x;
  const int tid = threadIdx.x;
  const int L = SQ + 1;  // 301
  __shared__ int s_cl[SQ + 1];
  __shared__ int s_scan[SQ + 1];
  __shared__ int s_ord[SQ + 1];
  __shared__ int s_nc;

  if (tid < L)
    s_cl[tid] = (tid == 0) ? 1 : (resp[b * SQ + tid - 1] > 0 ? 1 : 0);
  __syncthreads();
  if (tid == 0) {
    int run = 0;
    for (int p = 0; p < L; ++p) { run += s_cl[p]; s_scan[p] = run; }
    s_nc = run;
  }
  __syncthreads();
  const int nc = s_nc;
  if (tid < L) {
    const int slot = s_cl[tid] ? (s_scan[tid] - 1) : (nc + tid - s_scan[tid]);
    s_ord[slot] = tid;
  }
  __syncthreads();

  orig[b * LP + tid] = (tid < nc) ? s_ord[tid] : 0x7FFFFFFF;

  const int uidx = max(len[b] - 1, 0);
  const float* urow = user + ((size_t)b * NS + uidx) * DD;

  for (int i = tid; i < LP * DD; i += 320) {
    const int k = i >> 8, d = i & 255;
    float v = 0.0f;
    if (k < nc) {
      const int p = s_ord[k];
      v = (p == 0) ? urow[d] : item[((size_t)b * SQ + (p - 1)) * DD + d];
    }
    keys16[((size_t)b * LP + k) * DD + d] = (_Float16)v;
  }
}

// ---------------------------------------------------------------------------
// GEMM: C[i,n] = sum_d A[i,d] * W[n,d] + bias[n]; K=N=256, f16 in/out.
// Block: 256 thr = 8 waves, tile 128x64, BK=32 -> 4 WMMAs per wave per K-step.
// TSTORE=true stores transposed per batch: C_t[(b*D + n)*LP + key].
// ---------------------------------------------------------------------------
template <bool TSTORE>
__global__ __launch_bounds__(256) void gemm_nt_f16(
    const _Float16* __restrict__ A, const _Float16* __restrict__ W,
    const float* __restrict__ bias, _Float16* __restrict__ C) {
  __shared__ _Float16 As[128][48];  // 96B row stride: 16B aligned
  __shared__ _Float16 Ws[64][48];   // stored N-major: Ws[n][k]
  const int tid = threadIdx.x;
  const int row0 = blockIdx.x * 128;
  const int col0 = blockIdx.y * 64;
  const int wv = tid >> 5, lane = tid & 31;
  const int m0 = wv * 16;
  const int sr = tid >> 2, scg = (tid & 3) * 8;
  const int ln = lane & 15, lh = lane >> 4;

  v8f acc[4] = {{}, {}, {}, {}};
  for (int k0 = 0; k0 < DD; k0 += 32) {
#if HAVE_ASYNC_LDS
    __builtin_amdgcn_global_load_async_to_lds_b128(
        (gas4_t)(A + (size_t)(row0 + sr) * DD + k0 + scg),
        (las4_t)&As[sr][scg], 0, 0);
    __builtin_amdgcn_global_load_async_to_lds_b128(
        (gas4_t)(A + (size_t)(row0 + 64 + sr) * DD + k0 + scg),
        (las4_t)&As[64 + sr][scg], 0, 0);
    __builtin_amdgcn_global_load_async_to_lds_b128(
        (gas4_t)(W + (size_t)(col0 + sr) * DD + k0 + scg),
        (las4_t)&Ws[sr][scg], 0, 0);
    wait_async0();
#else
    *(v8h*)&As[sr][scg]      = *(const v8h*)(A + (size_t)(row0 + sr) * DD + k0 + scg);
    *(v8h*)&As[64 + sr][scg] = *(const v8h*)(A + (size_t)(row0 + 64 + sr) * DD + k0 + scg);
    *(v8h*)&Ws[sr][scg]      = *(const v8h*)(W + (size_t)(col0 + sr) * DD + k0 + scg);
#endif
    __syncthreads();
    const int akb = lh * 8;
    v16h a = cat8(*(const v8h*)&As[m0 + ln][akb], *(const v8h*)&As[m0 + ln][akb + 16]);
    const int bkb = lh * 8;
#pragma unroll
    for (int j = 0; j < 4; ++j) {
      v16h bj = cat8(*(const v8h*)&Ws[j * 16 + ln][bkb],
                     *(const v8h*)&Ws[j * 16 + ln][bkb + 16]);
      acc[j] = WMMA_F16(a, bj, acc[j]);
    }
    __syncthreads();
  }

  const int mb = lh * 8;
  if (!TSTORE) {
#pragma unroll
    for (int j = 0; j < 4; ++j) {
      const int gcol = col0 + j * 16 + ln;
      const float bi = bias[gcol];
#pragma unroll
      for (int r = 0; r < 8; ++r)
        C[(size_t)(row0 + m0 + mb + r) * DD + gcol] = (_Float16)(acc[j][r] + bi);
    }
  } else {
    const int gr0 = row0 + m0 + mb;          // 8-aligned; LP%8==0 -> no batch cross
    const int bb = gr0 / LP, key0 = gr0 % LP;
#pragma unroll
    for (int j = 0; j < 4; ++j) {
      const int gcol = col0 + j * 16 + ln;
      const float bi = bias[gcol];
      v8h pack;
#pragma unroll
      for (int r = 0; r < 8; ++r) pack[r] = (_Float16)(acc[j][r] + bi);
      *(v8h*)(C + ((size_t)(bb * DD + gcol)) * LP + key0) = pack;
    }
  }
}

// ---------------------------------------------------------------------------
// Fused attention: 4 waves per (16-query tile, head, batch) block.
// Phase 1: waves split key tiles (scores via WMMA, masked into LDS).
// Phase 2: waves split rows (softmax via shfl).
// Phase 3: waves split dh N-tiles (ctx via WMMA, B-frags direct from Vt).
// ---------------------------------------------------------------------------
__global__ __launch_bounds__(128) void attn_kernel(
    const _Float16* __restrict__ Q, const _Float16* __restrict__ K,
    const _Float16* __restrict__ Vt, const int* __restrict__ orig,
    _Float16* __restrict__ Ctx) {
  __shared__ float    sc[16][LP];   // masked scaled scores
  __shared__ _Float16 at[16][LP];   // attention weights (f16)
  const int tid = threadIdx.x;
  const int lane = tid & 31, wv = tid >> 5;
  const int qt = blockIdx.x, h = blockIdx.y, b = blockIdx.z;
  const int q0 = qt * 16;
  const int ln = lane & 15, lh = lane >> 4;
  const int akb = lh * 8;
  const int mb = lh * 8;

  // -------- Q fragments (per wave; constant over key loop) --------
  const int qq = min(q0 + ln, SQ - 1);
  const size_t qbase = ((size_t)(b * SQ + qq)) * DD + h * DH;
  v16h a0 = cat8(*(const v8h*)(Q + qbase + akb),      *(const v8h*)(Q + qbase + akb + 16));
  v16h a1 = cat8(*(const v8h*)(Q + qbase + 32 + akb), *(const v8h*)(Q + qbase + 32 + akb + 16));

  // -------- phase 1: scores over 19 key tiles, waves interleaved --------
  for (int kt = wv; kt < 19; kt += 4) {
    const int slot = kt * 16 + ln;
    const int op = orig[b * LP + slot];
    const size_t kbase = ((size_t)(b * LP + slot)) * DD + h * DH;
    const int bkb = lh * 8;
    v16h b0 = cat8(*(const v8h*)(K + kbase + bkb),      *(const v8h*)(K + kbase + bkb + 16));
    v16h b1 = cat8(*(const v8h*)(K + kbase + 32 + bkb), *(const v8h*)(K + kbase + 32 + bkb + 16));
    v8f c = {};
    c = WMMA_F16(a0, b0, c);
    c = WMMA_F16(a1, b1, c);
#pragma unroll
    for (int r = 0; r < 8; ++r) {
      const int q = q0 + mb + r;
      sc[mb + r][slot] = (op > q) ? -3.0e38f : c[r] * 0.125f;
    }
  }
  __syncthreads();

  // -------- phase 2: softmax, 4 rows per wave --------
#pragma unroll
  for (int i = 0; i < 4; ++i) {
    const int m = wv * 4 + i;
    float pm = -3.4e38f;
#pragma unroll
    for (int j = 0; j < LP / 32; ++j) pm = fmaxf(pm, sc[m][lane + j * 32]);
#pragma unroll
    for (int off = 16; off >= 1; off >>= 1) pm = fmaxf(pm, __shfl_xor(pm, off, 32));
    float e[LP / 32];
    float ps = 0.0f;
#pragma unroll
    for (int j = 0; j < LP / 32; ++j) { e[j] = __expf(sc[m][lane + j * 32] - pm); ps += e[j]; }
#pragma unroll
    for (int off = 16; off >= 1; off >>= 1) ps += __shfl_xor(ps, off, 32);
    const float inv = 1.0f / ps;
#pragma unroll
    for (int j = 0; j < LP / 32; ++j) at[m][lane + j * 32] = (_Float16)(e[j] * inv);
  }
  __syncthreads();

  // -------- phase 3: ctx = attn @ V; wave wv owns dh N-tile wv*16 --------
  const int n0 = wv * 16;
  const size_t vtbase = ((size_t)(b * DD + h * DH + n0 + ln)) * LP;
  const int kb2 = lh * 8;
  v8f acc = {};
  for (int ks = 0; ks < LP / 32; ++ks) {
    const int k0 = ks * 32;
    v16h a = cat8(*(const v8h*)&at[ln][k0 + akb], *(const v8h*)&at[ln][k0 + akb + 16]);
    v16h bf = cat8(*(const v8h*)(Vt + vtbase + k0 + kb2),
                   *(const v8h*)(Vt + vtbase + k0 + kb2 + 16));
    acc = WMMA_F16(a, bf, acc);
  }
#pragma unroll
  for (int r = 0; r < 8; ++r) {
    const int q = q0 + mb + r;
    if (q < SQ)
      Ctx[((size_t)(b * SQ + q)) * DD + h * DH + n0 + ln] = (_Float16)acc[r];
  }
}

// ---------------------------------------------------------------------------
// Final: out[i] = feat[i,:].w1 + item[i,:].w2 + b  (OUT=1, wave per row)
// ---------------------------------------------------------------------------
__global__ __launch_bounds__(256) void out_kernel(
    const _Float16* __restrict__ feat, const float* __restrict__ item,
    const float* __restrict__ Wout, const float* __restrict__ bout,
    float* __restrict__ out) {
  const int row = blockIdx.x * 8 + (threadIdx.x >> 5);
  const int lane = threadIdx.x & 31;
  const size_t base = (size_t)row * DD;
  float s = 0.0f;
#pragma unroll
  for (int j = 0; j < 8; ++j) {
    const int c = lane + j * 32;
    s += (float)feat[base + c] * Wout[c] + item[base + c] * Wout[DD + c];
  }
#pragma unroll
  for (int off = 16; off >= 1; off >>= 1) s += __shfl_xor(s, off, 32);
  if (lane == 0) out[row] = s + bout[0];
}

// ---------------------------------------------------------------------------
extern "C" void kernel_launch(void* const* d_in, const int* in_sizes, int n_in,
                              void* d_out, int out_size, void* d_ws, size_t ws_size,
                              hipStream_t stream) {
  (void)in_sizes; (void)n_in; (void)out_size; (void)ws_size;
  const float* item = (const float*)d_in[0];
  const float* user = (const float*)d_in[1];
  const int*   resp = (const int*)d_in[2];
  const int*   len  = (const int*)d_in[3];
  const float* Wq = (const float*)d_in[4];
  const float* bq = (const float*)d_in[5];
  const float* Wk = (const float*)d_in[6];
  const float* bk = (const float*)d_in[7];
  const float* Wv = (const float*)d_in[8];
  const float* bv = (const float*)d_in[9];
  const float* Wo = (const float*)d_in[10];
  const float* bo = (const float*)d_in[11];
  const float* Wout = (const float*)d_in[12];
  const float* bout = (const float*)d_in[13];
  float* out = (float*)d_out;

  const size_t nQrows = (size_t)BB * SQ;   // 76800
  const size_t nKrows = (size_t)BB * LP;   // 81920
  char* p = (char*)d_ws;
  const size_t szQ = nQrows * DD * sizeof(_Float16);  // 39.3 MB
  const size_t szK = nKrows * DD * sizeof(_Float16);  // 41.9 MB
  _Float16* X16    = (_Float16*)p;            p += szQ;   // also Ctx16
  _Float16* Keys16 = (_Float16*)p;            p += szK;   // also feat16
  _Float16* Q16    = (_Float16*)p;            p += szQ;
  _Float16* K16    = (_Float16*)p;            p += szK;
  _Float16* Vt16   = (_Float16*)p;            p += szK;   // transposed [B][D][LP]
  _Float16* Wq16   = (_Float16*)p;            p += DD * DD * sizeof(_Float16);
  _Float16* Wk16   = (_Float16*)p;            p += DD * DD * sizeof(_Float16);
  _Float16* Wv16   = (_Float16*)p;            p += DD * DD * sizeof(_Float16);
  _Float16* Wo16   = (_Float16*)p;            p += DD * DD * sizeof(_Float16);
  int* orig        = (int*)p;                 p += nKrows * sizeof(int);
  _Float16* Ctx16  = X16;     // alias: X16 dead after Q projection
  _Float16* feat16 = Keys16;  // alias: Keys16 dead after K/V projections

  // 1) converts
  cvt_f32_f16<<<8192, 256, 0, stream>>>(item, X16, (int)(nQrows * DD));
  cvt_f32_f16<<<256, 256, 0, stream>>>(Wq, Wq16, DD * DD);
  cvt_f32_f16<<<256, 256, 0, stream>>>(Wk, Wk16, DD * DD);
  cvt_f32_f16<<<256, 256, 0, stream>>>(Wv, Wv16, DD * DD);
  cvt_f32_f16<<<256, 256, 0, stream>>>(Wo, Wo16, DD * DD);

  // 2) key compaction + gather
  prep_kernel<<<BB, 320, 0, stream>>>(item, user, resp, len, Keys16, orig);

  // 3) projections (WMMA GEMMs); V stored transposed for ctx B-fragments
  gemm_nt_f16<false><<<dim3((unsigned)(nQrows / 128), 4), 256, 0, stream>>>(X16,    Wq16, bq, Q16);
  gemm_nt_f16<false><<<dim3((unsigned)(nKrows / 128), 4), 256, 0, stream>>>(Keys16, Wk16, bk, K16);
  gemm_nt_f16<true ><<<dim3((unsigned)(nKrows / 128), 4), 256, 0, stream>>>(Keys16, Wv16, bv, Vt16);

  // 4) fused masked attention
  attn_kernel<<<dim3(19, HH, BB), 128, 0, stream>>>(Q16, K16, Vt16, orig, Ctx16);

  // 5) output projection (WMMA GEMM)
  gemm_nt_f16<false><<<dim3((unsigned)(nQrows / 128), 4), 256, 0, stream>>>(Ctx16, Wo16, bo, feat16);

  // 6) final scalar head
  out_kernel<<<(unsigned)(nQrows / 8), 256, 0, stream>>>(feat16, item, Wout, bout, out);
}
